// Interaction_Estimator_31181462569635
// MI455X (gfx1250) — matrix operand order repaired and checked by reading, not compiled, not run on hardware
//
#include <hip/hip_runtime.h>
#include <stdint.h>

// ---------------------------------------------------------------------------
// Fused Interaction Estimator for MI455X (gfx1250, wave32)
//  - GEMMs run on v_wmma_f32_16x16x32_bf16 (fp32 accumulate)
//  - Weights pre-transposed+converted to bf16 in workspace (L2-resident)
//  - LayerNorm / ReLU / bilinear attention fused in-register, g_align staged
//    in LDS so the same 128 accumulator VGPRs serve both MLP blocks.
//  - VALU diet: packed f32->bf16 via v_perm_b32, sigmoid via v_exp/v_rcp
//    TRANS ops (co-execute with VALU) instead of full-precision divide.
// ---------------------------------------------------------------------------

typedef __attribute__((ext_vector_type(16))) __bf16 v16bf;
typedef __attribute__((ext_vector_type(8)))  float  v8f;

#define WAVES 4   // waves per block; LDS staging = WAVES * 16KB = 64KB

union Frag16 {
  v16bf v;
  unsigned d[8];
  uint4 q[2];
};

// Pack two f32 into one dword of bf16 (round-half-up, then v_perm_b32 gathers
// the two high halves: dst = { hi[31:16], lo[31:16] }).
static __device__ __forceinline__ unsigned pack_bf16(float lo, float hi) {
  unsigned a = __float_as_uint(lo) + 0x8000u;
  unsigned b = __float_as_uint(hi) + 0x8000u;
  return __builtin_amdgcn_perm(b, a, 0x07060302u);
}

static __device__ __forceinline__ unsigned short f2bf_rne(float f) {
  unsigned u = __float_as_uint(f);
  unsigned r = u + 0x7FFFu + ((u >> 16) & 1u);
  return (unsigned short)(r >> 16);
}

// Sum across the 16 lanes of a half-wave group (xor masks < 16 never cross
// the lane>>4 boundary, so rows M=v and M=v+8 reduce independently).
static __device__ __forceinline__ float hredu16(float x) {
  x += __shfl_xor(x, 1);
  x += __shfl_xor(x, 2);
  x += __shfl_xor(x, 4);
  x += __shfl_xor(x, 8);
  return x;
}

// sigmoid(x) = rcp(1 + exp2(-x*log2(e))) : 2 TRANS + 2 VALU ops
static __device__ __forceinline__ float sigmoidf_(float x) {
  float e = __builtin_amdgcn_exp2f(x * -1.4426950408889634f);
  return __builtin_amdgcn_rcpf(1.0f + e);
}

// One MLP block for a 16-row tile:  C = relu(LN(x @ W + b)) ; red = C . wa
// C[j][v] holds output element (row = v + 8*hi, col = 16*j + (lane&15)).
static __device__ __forceinline__ void mlp_block(
    const float* __restrict__ xrow,          // this lane's activation row
    const unsigned short* __restrict__ Wt,   // [256][256] bf16, Wt[d][k] = W[k][d]
    const float* __restrict__ bias,
    const float* __restrict__ gamma,
    const float* __restrict__ beta,
    const float* __restrict__ wa,
    int l15, int hi,
    v8f C[16], float red[8])
{
  v8f zero = {};
  #pragma unroll
  for (int j = 0; j < 16; ++j) C[j] = zero;

  for (int k0 = 0; k0 < 256; k0 += 32) {
    // A fragment (16x32 bf16): e -> K=k0+hi*16+e (matches B's contiguous K order)
    Frag16 A;
    const float4 a0 = *(const float4*)(xrow + k0 + hi * 16);
    const float4 a1 = *(const float4*)(xrow + k0 + hi * 16 + 4);
    const float4 a2 = *(const float4*)(xrow + k0 + hi * 16 + 8);
    const float4 a3 = *(const float4*)(xrow + k0 + hi * 16 + 12);
    A.d[0] = pack_bf16(a0.x, a0.y);
    A.d[1] = pack_bf16(a0.z, a0.w);
    A.d[2] = pack_bf16(a1.x, a1.y);
    A.d[3] = pack_bf16(a1.z, a1.w);
    A.d[4] = pack_bf16(a2.x, a2.y);
    A.d[5] = pack_bf16(a2.z, a2.w);
    A.d[6] = pack_bf16(a3.x, a3.y);
    A.d[7] = pack_bf16(a3.z, a3.w);

    #pragma unroll
    for (int j = 0; j < 16; ++j) {
      // B fragment (32x16 bf16): lane holds column N = 16*j + l15,
      // K = k0 + hi*16 .. +15 -> one contiguous 32B chunk of Wt[N][*].
      Frag16 Bf;
      const uint4* bp = (const uint4*)(Wt + (size_t)((j * 16 + l15) * 256 + k0 + hi * 16));
      Bf.q[0] = bp[0];
      Bf.q[1] = bp[1];
      C[j] = __builtin_amdgcn_wmma_f32_16x16x32_bf16(
          false, A.v, false, Bf.v, (short)0, C[j], false, false);
    }
  }

  // bias
  #pragma unroll
  for (int j = 0; j < 16; ++j) {
    float bb = bias[j * 16 + l15];
    #pragma unroll
    for (int v = 0; v < 8; ++v) C[j][v] += bb;
  }

  // LayerNorm stats per row (row = v + 8*hi lives in one 16-lane group)
  float mu[8], rs[8];
  #pragma unroll
  for (int v = 0; v < 8; ++v) {
    float s = 0.f, s2 = 0.f;
    #pragma unroll
    for (int j = 0; j < 16; ++j) { float t = C[j][v]; s += t; s2 = fmaf(t, t, s2); }
    s  = hredu16(s);
    s2 = hredu16(s2);
    float m   = s * (1.0f / 256.0f);
    float var = s2 * (1.0f / 256.0f) - m * m;
    mu[v] = m;
    rs[v] = rsqrtf(var + 1e-5f);
  }

  // normalize + affine + relu, and reduce against wa in the same pass
  #pragma unroll
  for (int v = 0; v < 8; ++v) red[v] = 0.f;
  #pragma unroll
  for (int j = 0; j < 16; ++j) {
    float gm = gamma[j * 16 + l15];
    float bt = beta[j * 16 + l15];
    float wv = wa[j * 16 + l15];
    #pragma unroll
    for (int v = 0; v < 8; ++v) {
      float y = fmaf((C[j][v] - mu[v]) * rs[v], gm, bt);
      y = fmaxf(y, 0.0f);
      C[j][v] = y;
      red[v] = fmaf(y, wv, red[v]);
    }
  }
  #pragma unroll
  for (int v = 0; v < 8; ++v) red[v] = hredu16(red[v]);
}

// Weight prep: Wt[sel][d][k] = bf16(W_sel[k][d])   (sel 0 = Wg, 1 = Wp)
__global__ void prep_w_kernel(const float* __restrict__ Wg,
                              const float* __restrict__ Wp,
                              unsigned short* __restrict__ Wt)
{
  int id = blockIdx.x * blockDim.x + threadIdx.x;   // 0 .. 131071
  int sel = id >> 16;
  int d   = (id >> 8) & 255;
  int k   = id & 255;
  const float* W = sel ? Wp : Wg;
  Wt[(sel << 16) + (d << 8) + k] = f2bf_rne(W[(k << 8) + d]);
}

__global__ __launch_bounds__(WAVES * 32)
void interaction_kernel(
    const float* __restrict__ gfeat, const float* __restrict__ pfeat,
    const unsigned short* __restrict__ WtG, const unsigned short* __restrict__ WtP,
    const float* __restrict__ bg, const float* __restrict__ g_ln_g, const float* __restrict__ g_ln_b,
    const float* __restrict__ bp, const float* __restrict__ p_ln_g, const float* __restrict__ p_ln_b,
    const float* __restrict__ wa_g, const float* __restrict__ ba_g,
    const float* __restrict__ wa_p, const float* __restrict__ ba_p,
    float* __restrict__ out, int rowtiles)
{
  __shared__ float lds_g[WAVES][16][256];   // 64 KB: g_align staging per wave

  const int lane = threadIdx.x & 31;
  const int wave = threadIdx.x >> 5;
  const int l15  = lane & 15;
  const int hi   = lane >> 4;
  const int tile = blockIdx.x * WAVES + wave;
  if (tile >= rowtiles) return;             // whole-wave uniform branch (EXEC stays all-ones)
  const int rowbase = tile * 16;

  v8f   C[16];
  float gred[8], pred[8];

  // ---- g branch: g_align = relu(LN(gfeat @ Wg + bg)); gred = g_align . wa_p
  const float* grow = gfeat + (size_t)(rowbase + l15) * 256;
  mlp_block(grow, WtG, bg, g_ln_g, g_ln_b, wa_p, l15, hi, C, gred);

  #pragma unroll
  for (int j = 0; j < 16; ++j) {
    #pragma unroll
    for (int v = 0; v < 8; ++v)
      lds_g[wave][v + 8 * hi][j * 16 + l15] = C[j][v];
  }

  // ---- p branch: p_align = relu(LN(pfeat @ Wp + bp)); pred = p_align . wa_g
  const float* prow = pfeat + (size_t)(rowbase + l15) * 256;
  mlp_block(prow, WtP, bp, p_ln_g, p_ln_b, wa_g, l15, hi, C, pred);

  const float bag = ba_g[0];
  const float bap = ba_p[0];

  // ---- combine: interaction = p*sigmoid(p*gred+ba_p) + g*sigmoid(g*pred+ba_g)
  #pragma unroll
  for (int j = 0; j < 16; ++j) {
    #pragma unroll
    for (int v = 0; v < 8; ++v) {
      float g = lds_g[wave][v + 8 * hi][j * 16 + l15];  // same-wave DS ops are in-order
      float p = C[j][v];
      float geno = sigmoidf_(fmaf(g, pred[v], bag));
      float path = sigmoidf_(fmaf(p, gred[v], bap));
      out[(size_t)(rowbase + v + 8 * hi) * 256 + j * 16 + l15] = fmaf(p, path, g * geno);
    }
  }
}

extern "C" void kernel_launch(void* const* d_in, const int* in_sizes, int n_in,
                              void* d_out, int out_size, void* d_ws, size_t ws_size,
                              hipStream_t stream)
{
  const float* gfeat = (const float*)d_in[0];
  const float* pfeat = (const float*)d_in[1];
  const float* Wg    = (const float*)d_in[2];
  const float* bg    = (const float*)d_in[3];
  const float* glng  = (const float*)d_in[4];
  const float* glnb  = (const float*)d_in[5];
  const float* Wp    = (const float*)d_in[6];
  const float* bpv   = (const float*)d_in[7];
  const float* plng  = (const float*)d_in[8];
  const float* plnb  = (const float*)d_in[9];
  const float* wag   = (const float*)d_in[10];
  const float* bag   = (const float*)d_in[11];
  const float* wap   = (const float*)d_in[12];
  const float* bap   = (const float*)d_in[13];
  float* out = (float*)d_out;

  // workspace: bf16 transposed weights, 2 * 256*256 * 2B = 256 KB
  unsigned short* Wt = (unsigned short*)d_ws;

  prep_w_kernel<<<512, 256, 0, stream>>>(Wg, Wp, Wt);

  int nrows    = in_sizes[0] / 256;        // B*L = 98304
  int rowtiles = nrows / 16;               // 6144 (divisible)
  int blocks   = (rowtiles + WAVES - 1) / WAVES;

  interaction_kernel<<<blocks, WAVES * 32, 0, stream>>>(
      gfeat, pfeat, Wt, Wt + 256 * 256,
      bg, glng, glnb, bpv, plng, plnb,
      wag, bag, wap, bap, out, rowtiles);
}
